// GCN_14671608283464
// MI455X (gfx1250) — compile-verified, matrix operand order checked by reading, not run on hardware
//
#include <hip/hip_runtime.h>
#include <cstdint>
#include <cstddef>

#define WG 256

typedef __attribute__((ext_vector_type(16))) __bf16 v16bf;
typedef __attribute__((ext_vector_type(8)))  __bf16 v8bf;
typedef __attribute__((ext_vector_type(8)))  float  v8f;

// fp32 -> bf16, round-to-nearest-even (pure bit ops; used only outside GEMM hot loops)
__device__ __forceinline__ __bf16 f2bf(float f) {
    union { float f; unsigned u; } c; c.f = f;
    unsigned u = c.u;
    u += 0x7FFFu + ((u >> 16) & 1u);
    union { unsigned short s; __bf16 b; } o; o.s = (unsigned short)(u >> 16);
    return o.b;
}

__device__ __forceinline__ void atomAddF(float* p, float v) {
    __hip_atomic_fetch_add(p, v, __ATOMIC_RELAXED, __HIP_MEMORY_SCOPE_AGENT);
}

// ---------------- prep kernels ----------------
__global__ void k_fill(float* __restrict__ p, int n, float v) {
    int i = blockIdx.x * blockDim.x + threadIdx.x;
    if (i < n) p[i] = v;
}

// packed fp32 -> bf16 conversion, 4 elems/thread
__global__ void k_cvtbf(const float* __restrict__ in, __bf16* __restrict__ out, int n4) {
    int i = blockIdx.x * blockDim.x + threadIdx.x;
    if (i >= n4) return;
    int base = i * 4;
    float4 v = *(const float4*)(in + base);
    union { __bf16 b[4]; uint2 u; } pk;
    pk.b[0] = f2bf(v.x); pk.b[1] = f2bf(v.y); pk.b[2] = f2bf(v.z); pk.b[3] = f2bf(v.w);
    *(uint2*)(out + base) = pk.u;
}

// Bt[n][k] = bf16(W[k][n]); rows n >= Nout zero-padded
__global__ void k_transpose_bf16(const float* __restrict__ W, __bf16* __restrict__ Bt,
                                 int K, int Nout, int NoutPad) {
    int i = blockIdx.x * blockDim.x + threadIdx.x;
    if (i >= K * NoutPad) return;
    int n = i / K, k = i - n * K;
    float v = (n < Nout) ? W[(size_t)k * Nout + n] : 0.0f;
    Bt[(size_t)n * K + k] = f2bf(v);
}

__global__ void k_padbias(const float* __restrict__ b, float* __restrict__ bp, int n, int npad) {
    int i = threadIdx.x;
    if (i < npad) bp[i] = (i < n) ? b[i] : 0.0f;
}

__global__ void k_degcount(const int* __restrict__ ei, float* __restrict__ deg, int E) {
    int e = blockIdx.x * blockDim.x + threadIdx.x;
    if (e < E) atomAddF(&deg[ei[E + e]], 1.0f);
}

__global__ void k_dinv(const float* __restrict__ deg, float* __restrict__ dinv, int n) {
    int i = blockIdx.x * blockDim.x + threadIdx.x;
    if (i < n) dinv[i] = rsqrtf(deg[i] + 1.0f);
}

// ---------------- WMMA GEMM ----------------
// C[M,Nout] = A[M,K](bf16) * Bt[Nout,K](bf16)^T (+bias)
// One wave computes a (16*MT) x (16*NT) tile. Per the CDNA5 16-bit fragment layout,
// lane (l&15) selects the row/col and (l>>4) selects K-halves, so each fragment is
// two contiguous 16B loads (global_load_b128) from bf16 row-major storage.
// B fragments are shared across the MT row tiles -> 2*MT + 2*NT loads per 32-K step
// feeding MT*NT v_wmma_f32_16x16x32_bf16.
template<int MT, int NT, bool OUTBF>
__global__ __launch_bounds__(WG)
void k_gemm_bf16(const __bf16* __restrict__ A, const __bf16* __restrict__ Bt,
                 const float* __restrict__ bias, void* __restrict__ Cv,
                 int M, int K, int Nout, int numTileCols) {
    int waveId  = (int)((blockIdx.x * (unsigned)blockDim.x + threadIdx.x) >> 5);
    int lane    = threadIdx.x & 31;
    int tileRow = waveId / numTileCols;
    int tcg     = waveId - tileRow * numTileCols;
    if (tileRow * (16 * MT) >= M) return;

    int idx = lane & 15;
    int hi  = lane >> 4;
    int nbase = tcg * (16 * NT);

    const __bf16* Ar[MT];
#pragma unroll
    for (int m = 0; m < MT; m++)
        Ar[m] = A + (size_t)(tileRow * (16 * MT) + m * 16 + idx) * K;
    const __bf16* Br[NT];
#pragma unroll
    for (int t = 0; t < NT; t++)
        Br[t] = Bt + (size_t)(nbase + t * 16 + idx) * K;

    v8f acc[MT][NT] = {};

    for (int k = 0; k < K; k += 32) {
        int ka = k + hi * 8;     // elems 0..7  <- K = ka..ka+7
        int kb = ka + 16;        // elems 8..15 <- K = kb..kb+7
        v16bf a[MT];
#pragma unroll
        for (int m = 0; m < MT; m++) {
            v8bf a0 = *(const v8bf*)(Ar[m] + ka);
            v8bf a1 = *(const v8bf*)(Ar[m] + kb);
            a[m] = __builtin_shufflevector(a0, a1, 0,1,2,3,4,5,6,7,8,9,10,11,12,13,14,15);
        }
#pragma unroll
        for (int t = 0; t < NT; t++) {
            v8bf b0 = *(const v8bf*)(Br[t] + ka);
            v8bf b1 = *(const v8bf*)(Br[t] + kb);
            v16bf b = __builtin_shufflevector(b0, b1, 0,1,2,3,4,5,6,7,8,9,10,11,12,13,14,15);
#pragma unroll
            for (int m = 0; m < MT; m++)
                acc[m][t] = __builtin_amdgcn_wmma_f32_16x16x32_bf16(
                    false, a[m], false, b, (short)0, acc[m][t], false, false);
        }
    }

    // C/D layout: VGPR r -> local row r (+8 for lanes 16-31), local col = idx
#pragma unroll
    for (int t = 0; t < NT; t++) {
        int col = nbase + t * 16 + idx;
        float bv = bias ? bias[col] : 0.0f;
#pragma unroll
        for (int m = 0; m < MT; m++) {
#pragma unroll
            for (int r = 0; r < 8; r++) {
                int row = tileRow * (16 * MT) + m * 16 + hi * 8 + r;
                float v = acc[m][t][r] + bv;
                if (OUTBF) ((__bf16*)Cv)[(size_t)row * Nout + col] = f2bf(v);
                else       ((float*) Cv)[(size_t)row * Nout + col] = v;
            }
        }
    }
}

// ---------------- graph aggregation ----------------
// G[i][f] = T[i][f]*dinv[i]^2 + b[f]   (self-loop term + bias), 4 elems/thread
__global__ void k_selfinit(const float* __restrict__ T, const float* __restrict__ dinv,
                           const float* __restrict__ b, float* __restrict__ G, int n, int F) {
    int i = blockIdx.x * blockDim.x + threadIdx.x;
    int base = i * 4;
    if (base >= n * F) return;
    int r = base / F, f = base - r * F;
    float d = dinv[r]; d *= d;
    float4 t = *(const float4*)(T + base);
    float4 o;
    o.x = t.x * d + b[f];
    o.y = t.y * d + b[f + 1];
    o.z = t.z * d + b[f + 2];
    o.w = t.w * d + b[f + 3];
    *(float4*)(G + base) = o;
}

// one wave32 per edge; each lane owns F/32 features -> global_atomic_add_f32
__global__ void k_scatter(const int* __restrict__ ei, const float* __restrict__ T,
                          const float* __restrict__ dinv, float* __restrict__ G,
                          int E, int F) {
    int wave = (int)((blockIdx.x * (unsigned)blockDim.x + threadIdx.x) >> 5);
    int lane = threadIdx.x & 31;
    if (wave >= E) return;
    int src = ei[wave], dst = ei[E + wave];
    float c = dinv[src] * dinv[dst];
    const float* Ts = T + (size_t)src * F;
    float* Gd = G + (size_t)dst * F;
    int per = F >> 5;
    int f0 = lane * per;
    for (int j = 0; j < per; j++)
        atomAddF(&Gd[f0 + j], Ts[f0 + j] * c);
}

// BN(eval)+ReLU, output as bf16 (consumed only as the next GEMM's A operand)
__global__ void k_bnrelu_bf(const float* __restrict__ G, const float* __restrict__ g,
                            const float* __restrict__ be, const float* __restrict__ mu,
                            const float* __restrict__ va, __bf16* __restrict__ out,
                            int n, int F) {
    int i = blockIdx.x * blockDim.x + threadIdx.x;
    int base = i * 4;
    if (base >= n * F) return;
    int f = base % F;
    float4 x = *(const float4*)(G + base);
    union { __bf16 b[4]; uint2 u; } pk;
    float y0 = (x.x - mu[f])     * rsqrtf(va[f]     + 1e-5f) * g[f]     + be[f];
    float y1 = (x.y - mu[f + 1]) * rsqrtf(va[f + 1] + 1e-5f) * g[f + 1] + be[f + 1];
    float y2 = (x.z - mu[f + 2]) * rsqrtf(va[f + 2] + 1e-5f) * g[f + 2] + be[f + 2];
    float y3 = (x.w - mu[f + 3]) * rsqrtf(va[f + 3] + 1e-5f) * g[f + 3] + be[f + 3];
    pk.b[0] = f2bf(y0 > 0.0f ? y0 : 0.0f);
    pk.b[1] = f2bf(y1 > 0.0f ? y1 : 0.0f);
    pk.b[2] = f2bf(y2 > 0.0f ? y2 : 0.0f);
    pk.b[3] = f2bf(y3 > 0.0f ? y3 : 0.0f);
    *(uint2*)(out + base) = pk.u;
}

__global__ void k_logsoftmax(const float* __restrict__ L, float* __restrict__ out,
                             int n, int Fp, int F) {
    int i = blockIdx.x * blockDim.x + threadIdx.x;
    if (i >= n) return;
    const float* row = L + (size_t)i * Fp;
    float m = row[0];
    for (int j = 1; j < F; j++) m = fmaxf(m, row[j]);
    float s = 0.0f;
    for (int j = 0; j < F; j++) s += __expf(row[j] - m);
    float ls = __logf(s);
    for (int j = 0; j < F; j++) out[(size_t)i * F + j] = row[j] - m - ls;
}

// ---------------- launcher ----------------
extern "C" void kernel_launch(void* const* d_in, const int* in_sizes, int n_in,
                              void* d_out, int out_size, void* d_ws, size_t ws_size,
                              hipStream_t stream) {
    (void)n_in; (void)out_size; (void)ws_size;
    const float* x    = (const float*)d_in[0];
    const int*   ei   = (const int*)d_in[1];
    const float* W0   = (const float*)d_in[2];
    const float* b0   = (const float*)d_in[3];
    const float* W1   = (const float*)d_in[4];
    const float* b1   = (const float*)d_in[5];
    const float* g1   = (const float*)d_in[6];
    const float* be1  = (const float*)d_in[7];
    const float* m1   = (const float*)d_in[8];
    const float* v1   = (const float*)d_in[9];
    const float* W2   = (const float*)d_in[10];
    const float* b2   = (const float*)d_in[11];
    const float* g2   = (const float*)d_in[12];
    const float* be2  = (const float*)d_in[13];
    const float* m2   = (const float*)d_in[14];
    const float* v2   = (const float*)d_in[15];
    const float* Wout = (const float*)d_in[16];
    const float* bout = (const float*)d_in[17];

    const int N = 100000, IN = 512, F0 = 512, HID = 256, NC = 20, NCP = 32;
    const int E = in_sizes[1] / 2;

    // workspace carve-up (256B aligned)
    char* base = (char*)d_ws;
    size_t off = 0;
    auto alloc = [&](size_t bytes) -> char* {
        char* p = base + off;
        off = (off + bytes + 255) & ~(size_t)255;
        return p;
    };
    float*  deg  = (float*) alloc((size_t)N * 4);
    float*  dinv = (float*) alloc((size_t)N * 4);
    __bf16* W0t  = (__bf16*)alloc((size_t)F0  * IN  * 2);
    __bf16* W1t  = (__bf16*)alloc((size_t)HID * F0  * 2);
    __bf16* W2t  = (__bf16*)alloc((size_t)HID * HID * 2);
    __bf16* Wot  = (__bf16*)alloc((size_t)NCP * HID * 2);
    float*  bop  = (float*) alloc((size_t)NCP * 4);
    __bf16* Xbf  = (__bf16*)alloc((size_t)N * IN  * 2);  // bf16 copy of x
    __bf16* H0b  = (__bf16*)alloc((size_t)N * F0  * 2);  // lin0 output (bf16, GEMM-A only)
    float*  T    = (float*) alloc((size_t)N * HID * 4);  // GEMM f32 temp; reused for logits
    float*  G    = (float*) alloc((size_t)N * HID * 4);  // aggregation buffer (both convs)
    __bf16* G1b  = (__bf16*)alloc((size_t)N * HID * 2);  // post-BN1 (bf16, GEMM-A only)
    __bf16* G2b  = (__bf16*)alloc((size_t)N * HID * 2);  // post-BN2 (bf16, GEMM-A only)

    auto cdiv = [](long long a, long long b) { return (int)((a + b - 1) / b); };

    // --- prep ---
    k_transpose_bf16<<<cdiv((long long)IN * F0,  WG), WG, 0, stream>>>(W0,   W0t, IN,  F0,  F0);
    k_transpose_bf16<<<cdiv((long long)F0 * HID, WG), WG, 0, stream>>>(W1,   W1t, F0,  HID, HID);
    k_transpose_bf16<<<cdiv((long long)HID* HID, WG), WG, 0, stream>>>(W2,   W2t, HID, HID, HID);
    k_transpose_bf16<<<cdiv((long long)HID* NCP, WG), WG, 0, stream>>>(Wout, Wot, HID, NC,  NCP);
    k_padbias<<<1, NCP, 0, stream>>>(bout, bop, NC, NCP);
    k_cvtbf<<<cdiv((long long)N * IN / 4, WG), WG, 0, stream>>>(x, Xbf, N * IN / 4);
    k_fill<<<cdiv(N, WG), WG, 0, stream>>>(deg, N, 0.0f);
    k_degcount<<<cdiv(E, WG), WG, 0, stream>>>(ei, deg, E);
    k_dinv<<<cdiv(N, WG), WG, 0, stream>>>(deg, dinv, N);

    // --- lin0: H0b = bf16(x @ W0 + b0) ---
    {
        int cols = F0 / 64;                       // 8
        long long waves = (long long)(N / 32) * cols;
        k_gemm_bf16<2, 4, true><<<cdiv(waves * 32, WG), WG, 0, stream>>>(
            Xbf, W0t, b0, H0b, N, IN, F0, cols);
    }
    // --- conv1: T = H0 @ W1 ; G = self + scatter ; BN+ReLU -> G1b ---
    {
        int cols = HID / 64;                      // 4
        long long waves = (long long)(N / 32) * cols;
        k_gemm_bf16<2, 4, false><<<cdiv(waves * 32, WG), WG, 0, stream>>>(
            H0b, W1t, nullptr, T, N, F0, HID, cols);
    }
    k_selfinit <<<cdiv((long long)N * HID / 4, WG), WG, 0, stream>>>(T, dinv, b1, G, N, HID);
    k_scatter  <<<cdiv((long long)E * 32,      WG), WG, 0, stream>>>(ei, T, dinv, G, E, HID);
    k_bnrelu_bf<<<cdiv((long long)N * HID / 4, WG), WG, 0, stream>>>(G, g1, be1, m1, v1, G1b, N, HID);
    // --- conv2 ---
    {
        int cols = HID / 64;
        long long waves = (long long)(N / 32) * cols;
        k_gemm_bf16<2, 4, false><<<cdiv(waves * 32, WG), WG, 0, stream>>>(
            G1b, W2t, nullptr, T, N, HID, HID, cols);
    }
    k_selfinit <<<cdiv((long long)N * HID / 4, WG), WG, 0, stream>>>(T, dinv, b2, G, N, HID);
    k_scatter  <<<cdiv((long long)E * 32,      WG), WG, 0, stream>>>(ei, T, dinv, G, E, HID);
    k_bnrelu_bf<<<cdiv((long long)N * HID / 4, WG), WG, 0, stream>>>(G, g2, be2, m2, v2, G2b, N, HID);
    // --- logits (padded Nout=32) into T, then log_softmax over 20 ---
    {
        long long waves = N / 32;
        k_gemm_bf16<2, 2, false><<<cdiv(waves * 32, WG), WG, 0, stream>>>(
            G2b, Wot, bop, T, N, HID, NCP, 1);
    }
    k_logsoftmax<<<cdiv(N, WG), WG, 0, stream>>>(T, (float*)d_out, N, NCP, NC);
}
